// SimpleRecurrentCell_32710470926766
// MI455X (gfx1250) — matrix operand order repaired and checked by reading, not compiled, not run on hardware
//
#include <hip/hip_runtime.h>
#include <math.h>

// ---------------------------------------------------------------------------
// LSTM rollout (576 sequential steps) as a single persistent kernel on
// gfx1250. Per step: gates = h @ W^T (+x*w_ih +bias) via v_wmma_f32_16x16x32_bf16,
// cell update in registers, h written bf16 (ping-pong), device-wide barrier.
// Future steps fold x = h.Wdec + bdec into W_eff (FEAT==1 rank-1 update).
// ---------------------------------------------------------------------------

#define B_SZ   256
#define HID    512
#define G4H    2048
#define PAST_T 512
#define TOT_T  576
#define NWG    64
#define TPB    256      // 8 wave32 per WG -> 512 waves total

typedef __attribute__((ext_vector_type(16))) __bf16   v16bf;
typedef __attribute__((ext_vector_type(8)))  float    v8f;
typedef __attribute__((ext_vector_type(4)))  unsigned uv4;

union Frag { v16bf v; uv4 q[2]; };

__device__ __forceinline__ unsigned short f2bf(float f) {
  unsigned u = __float_as_uint(f);
  u += 0x7FFFu + ((u >> 16) & 1u);          // round-to-nearest-even
  return (unsigned short)(u >> 16);
}
__device__ __forceinline__ float bf2f(unsigned short s) {
  return __uint_as_float(((unsigned)s) << 16);
}
__device__ __forceinline__ float sigm(float x) { return 1.0f / (1.0f + __expf(-x)); }

// ---- workspace layout (bytes), all 256-aligned; total ~5.03 MB ------------
#define OFF_CTRL 0u            // 2 x u32 barrier state
#define OFF_W    256u          // bf16 weights [2][2048][512]      = 4194304 B
#define OFF_BIAS 4194560u      // f32 [2][2048]                    =   16384 B
#define OFF_WIH  4210944u      // f32 [2048]  (W_ih column)        =    8192 B
#define OFF_WDEC 4219136u      // f32 [512]                        =    2048 B
#define OFF_C    4221184u      // f32 [256][512] cell state        =  524288 B
#define OFF_H    4745472u      // bf16 [2][256][512] h ping-pong   =  524288 B

// ---------------------------------------------------------------------------
__global__ void lstm_init(const float* __restrict__ W_ih,
                          const float* __restrict__ b_ih,
                          const float* __restrict__ b_hh,
                          const float* __restrict__ W_dec,
                          const float* __restrict__ b_dec,
                          float* __restrict__ biasArr, float* __restrict__ wih0,
                          float* __restrict__ wdec, float* __restrict__ c_state,
                          unsigned short* __restrict__ h0, unsigned* __restrict__ ctrl) {
  int i = blockIdx.x * blockDim.x + threadIdx.x;
  if (i < B_SZ * HID) {
    c_state[i] = 0.01f;
    h0[i] = f2bf(0.01f);
  }
  if (i < G4H) {
    float b0 = b_ih[i] + b_hh[i];
    biasArr[i] = b0;
    biasArr[G4H + i] = b0 + b_dec[0] * W_ih[i];   // folded future bias
    wih0[i] = W_ih[i];
  }
  if (i < HID) wdec[i] = W_dec[i];
  if (i == 0) { ctrl[0] = 0u; ctrl[1] = 0u; }
}

// variant 0: W_hh   variant 1: W_hh + W_ih[:,0] (outer) W_dec[0,:]
__global__ void lstm_pack(const float* __restrict__ W_hh,
                          const float* __restrict__ W_ih,
                          const float* __restrict__ W_dec,
                          unsigned short* __restrict__ Wbf) {
  int i = blockIdx.x * blockDim.x + threadIdx.x;
  if (i >= 2 * G4H * HID) return;
  int variant = i / (G4H * HID);
  int r = i - variant * (G4H * HID);
  int n = r / HID;
  int k = r - n * HID;
  float v = W_hh[n * HID + k];
  if (variant) v += W_ih[n] * W_dec[k];
  Wbf[i] = f2bf(v);
}

// ---------------------------------------------------------------------------
__device__ __forceinline__ void grid_barrier(unsigned* cnt, unsigned* gen,
                                             unsigned* lgen) {
  __threadfence();                 // release this WG's global stores
  __syncthreads();
  if (threadIdx.x == 0) {
    unsigned g = *lgen;
    if (atomicAdd(cnt, 1u) == (unsigned)(NWG - 1)) {
      *(volatile unsigned*)cnt = 0u;
      __threadfence();
      atomicAdd(gen, 1u);
    } else {
      while (*(volatile unsigned*)gen == g) __builtin_amdgcn_s_sleep(1);
    }
    *lgen = g + 1u;
  }
  __syncthreads();
  __threadfence();                 // acquire others' stores
}

__device__ __forceinline__ void decode_row(const unsigned short* __restrict__ h,
                                           const float* __restrict__ wdec,
                                           float bdec, float* __restrict__ out,
                                           int b, int tcol, int lane) {
  float s = 0.0f;
  for (int k = lane; k < HID; k += 32)
    s += bf2f(h[b * HID + k]) * wdec[k];
  for (int off = 16; off; off >>= 1) s += __shfl_xor(s, off, 32);
  if (lane == 0) out[b * TOT_T + tcol] = s + bdec;
}

// ---------------------------------------------------------------------------
__global__ __launch_bounds__(TPB) void lstm_persistent(
    const float* __restrict__ x_in,        // input_seq [256][512][1]
    const float* __restrict__ b_dec_p,
    const unsigned short* __restrict__ Wbf,   // [2][2048][512] bf16
    const float* __restrict__ biasArr,        // [2][2048]
    const float* __restrict__ wih0,           // [2048]
    const float* __restrict__ wdec,           // [512]
    float* __restrict__ c_state,              // [256][512]
    unsigned short* __restrict__ hbuf,        // [2][256][512] bf16
    unsigned* __restrict__ ctrl,
    float* __restrict__ out)                  // [256][576]
{
  const int lane  = threadIdx.x & 31;
  const int wid   = (blockIdx.x * TPB + threadIdx.x) >> 5;  // 0..511
  const int m0    = (wid >> 5) * 16;     // batch tile base
  const int n0    = (wid & 31) * 16;     // hidden tile base (within HID)
  const int lhalf = lane & 15;
  const int upper = lane >> 4;           // 0: lanes 0-15, 1: lanes 16-31
  const float bdec = b_dec_p[0];
  unsigned lgen = 0;

#pragma unroll 1
  for (int t = 0; t < TOT_T; ++t) {
    const unsigned short* hcur = hbuf + (t & 1) * (B_SZ * HID);
    unsigned short*       hnxt = hbuf + ((t & 1) ^ 1) * (B_SZ * HID);
    const int past = (t < PAST_T);
    const unsigned short* W    = Wbf + (past ? 0 : (G4H * HID));
    const float*          bias = biasArr + (past ? 0 : G4H);

    // decode previous step's h (h_t) into out[:, t-1]; pipelined, no extra sync
    if (t > 0 && wid < B_SZ)
      decode_row(hcur, wdec, bdec, out, wid, t - 1, lane);

    // ---- gates[16x16 x 4] = h_tile @ W^T ---------------------------------
    v8f acc[4] = {};
    // A fragment base: lane<16 -> K {0..7,16..23}; lane>=16 -> +8
    const unsigned short* arow = hcur + (m0 + lhalf) * HID + upper * 8;
    const int Ncol = n0 + lhalf;
    // B fragment base: lane<16 -> K {0..7,16..23}; lane>=16 -> +8 (row-major W)
    const unsigned short* brow = W + (size_t)Ncol * HID + upper * 8;

    for (int k0 = 0; k0 < HID; k0 += 32) {
      Frag a;
      a.q[0] = *(const uv4*)(arow + k0);
      a.q[1] = *(const uv4*)(arow + k0 + 16);
#pragma unroll
      for (int g = 0; g < 4; ++g) {
        Frag bf;
        const unsigned short* bp = brow + (size_t)g * (512 * HID) + k0;
        bf.q[0] = *(const uv4*)(bp);
        bf.q[1] = *(const uv4*)(bp + 16);
        acc[g] = __builtin_amdgcn_wmma_f32_16x16x32_bf16(
            false, a.v, false, bf.v, (short)0, acc[g], false, false);
      }
    }

    // ---- cell update in registers (C/D layout: VGPR r, lane half -> M) ---
    float bi = bias[Ncol],           bfv = bias[HID + Ncol];
    float bg = bias[2 * HID + Ncol], bo  = bias[3 * HID + Ncol];
    float wi = 0.f, wf = 0.f, wg = 0.f, wo = 0.f;
    if (past) {
      wi = wih0[Ncol];            wf = wih0[HID + Ncol];
      wg = wih0[2 * HID + Ncol];  wo = wih0[3 * HID + Ncol];
    }

#pragma unroll
    for (int r = 0; r < 8; ++r) {
      int bidx = m0 + r + upper * 8;
      float xv = past ? x_in[bidx * PAST_T + t] : 0.0f;
      float gi = acc[0][r] + bi  + xv * wi;
      float gf = acc[1][r] + bfv + xv * wf;
      float gg = acc[2][r] + bg  + xv * wg;
      float go = acc[3][r] + bo  + xv * wo;
      float c_old = c_state[bidx * HID + Ncol];
      float cN = sigm(gf) * c_old + sigm(gi) * tanhf(gg);
      c_state[bidx * HID + Ncol] = cN;
      float hN = sigm(go) * tanhf(cN);
      hnxt[bidx * HID + Ncol] = f2bf(hN);
    }

    grid_barrier(&ctrl[0], &ctrl[1], &lgen);
  }

  // final hidden state h_576 lives in buffer 0 -> out[:, 575]
  if (wid < B_SZ)
    decode_row(hbuf, wdec, bdec, out, wid, TOT_T - 1, lane);
}

// ---------------------------------------------------------------------------
extern "C" void kernel_launch(void* const* d_in, const int* in_sizes, int n_in,
                              void* d_out, int out_size, void* d_ws, size_t ws_size,
                              hipStream_t stream) {
  (void)in_sizes; (void)n_in; (void)out_size; (void)ws_size;
  const float* input_seq = (const float*)d_in[0];
  // d_in[1] = future_n (compile-time 64 here)
  const float* W_ih  = (const float*)d_in[2];
  const float* W_hh  = (const float*)d_in[3];
  const float* b_ih  = (const float*)d_in[4];
  const float* b_hh  = (const float*)d_in[5];
  const float* W_dec = (const float*)d_in[6];
  const float* b_dec = (const float*)d_in[7];
  float* out = (float*)d_out;

  char* ws = (char*)d_ws;
  unsigned*       ctrl    = (unsigned*)(ws + OFF_CTRL);
  unsigned short* Wbf     = (unsigned short*)(ws + OFF_W);
  float*          biasArr = (float*)(ws + OFF_BIAS);
  float*          wih0    = (float*)(ws + OFF_WIH);
  float*          wdec    = (float*)(ws + OFF_WDEC);
  float*          c_state = (float*)(ws + OFF_C);
  unsigned short* hbuf    = (unsigned short*)(ws + OFF_H);

  lstm_init<<<(B_SZ * HID + TPB - 1) / TPB, TPB, 0, stream>>>(
      W_ih, b_ih, b_hh, W_dec, b_dec, biasArr, wih0, wdec, c_state, hbuf, ctrl);

  lstm_pack<<<(2 * G4H * HID + TPB - 1) / TPB, TPB, 0, stream>>>(
      W_hh, W_ih, W_dec, Wbf);

  lstm_persistent<<<NWG, TPB, 0, stream>>>(
      input_seq, b_dec, Wbf, biasArr, wih0, wdec, c_state, hbuf, ctrl, out);
}
